// Bottleneck_88888643158615
// MI455X (gfx1250) — 8-bit matrix operand order repaired and checked by reading, compiled, not run on hardware
//
#include <hip/hip_runtime.h>


typedef __attribute__((ext_vector_type(8))) int v8i;

#define LDS_STRIDE 80


__global__ __launch_bounds__(256)
void pack_x_kernel(const int* __restrict__ x, signed char* __restrict__ x8) {
  size_t idx = (size_t)blockIdx.x * 256 + threadIdx.x;
  int c   = (int)(idx & 1023);
  size_t q = idx >> 10;
  int hw  = (int)(q % 784);
  int n   = (int)(q / 784);
  x8[idx] = (signed char)x[((size_t)n * 1024 + c) * 784 + hw];
}

__global__ __launch_bounds__(256)
void pack_w11_kernel(const float* __restrict__ w2, const int* __restrict__ s,
                     signed char* __restrict__ w) {
  size_t idx = (size_t)blockIdx.x * 256 + threadIdx.x;
  int v = (int)w2[idx];
  w[idx] = (signed char)(v << s[idx]);
}

__global__ __launch_bounds__(256)
void pack_w33_kernel(const float* __restrict__ w2, const int* __restrict__ s,
                     signed char* __restrict__ wt) {
  size_t idx = (size_t)blockIdx.x * 256 + threadIdx.x;
  int ic  = (int)(idx & 255);
  size_t q = idx >> 8;
  int oc  = (int)(q & 255);
  int tap = (int)(q >> 8);
  size_t src = ((size_t)oc * 256 + ic) * 9 + tap;
  int v = (int)w2[src];
  wt[idx] = (signed char)(v << s[src]);
}

__global__
void prep_bn_kernel(const float* __restrict__ alpha, const float* __restrict__ beta,
                    const int* __restrict__ q, float* __restrict__ bnA,
                    float* __restrict__ bnB, int C) {
  int i = blockIdx.x * 256 + threadIdx.x;
  if (i < C) {
    bnA[i] = alpha[i] * (1.0f / 4096.0f);
    bnB[i] = beta[i] * exp2f((float)q[i]);
  }
}

__global__ __launch_bounds__(256)
void zero16_kernel(int4* __restrict__ p) {
  size_t idx = (size_t)blockIdx.x * 256 + threadIdx.x;
  int4 z; z.x = 0; z.y = 0; z.z = 0; z.w = 0;
  p[idx] = z;
}

template <int MODE>
__global__ __launch_bounds__(256)
void gemm_bn_kernel(const signed char* __restrict__ Aw,
                    const signed char* __restrict__ Bact,
                    signed char* __restrict__ out8,
                    float* __restrict__ outf,
                    const int* __restrict__ xid,
                    const float* __restrict__ bnA,
                    const float* __restrict__ bnB) {
  __shared__ signed char At[128 * LDS_STRIDE];
  __shared__ signed char Bt[128 * LDS_STRIDE];

  const int ntile = blockIdx.x * 128;
  const int mtile = blockIdx.y * 128;
  const int t    = threadIdx.x;
  const int lane = t & 31;
  const int wave = t >> 5;
  const int wm   = wave & 3;
  const int wn   = wave >> 2;
  const int l15  = lane & 15;
  const int hi8  = (lane >> 4) << 3;
  const int hi16 = (lane >> 4) << 4;

  v8i acc[2][4] = {};

  const int KSTEPS = (MODE == 1) ? 16 : ((MODE == 2) ? 36 : 4);
  const int ASTRIDE = (MODE == 1) ? 1024 : 256;

  for (int ks = 0; ks < KSTEPS; ++ks) {
    const int tap = (MODE == 2) ? (ks >> 2) : 0;
    const int kc  = (MODE == 2) ? (ks & 3) : ks;
    const int k0  = kc * 64;

    __syncthreads();
    {
      const int i = t >> 1, part = t & 1;
      const signed char* src;
      if (MODE == 2)
        src = Aw + (((size_t)tap * 256 + (mtile + i)) * 256) + k0 + part * 32;
      else
        src = Aw + (size_t)(mtile + i) * ASTRIDE + k0 + part * 32;
      const int4* g = (const int4*)src;
      int4 d0 = g[0];
      int4 d1 = g[1];
      *(int4*)&At[i * LDS_STRIDE + part * 32 + 0]  = d0;
      *(int4*)&At[i * LDS_STRIDE + part * 32 + 16] = d1;
    }
    {
      const int i = t >> 1, part = t & 1;
      const int p = ntile + i;
      size_t boff;
      if (MODE == 1) {
        boff = (size_t)p * 1024 + k0;
      } else if (MODE == 3) {
        boff = (size_t)p * 256 + k0;
      } else {
        const int n = p / 784, rem = p % 784, h = rem / 28, w = rem % 28;
        const int ky = tap / 3, kx = tap % 3;
        boff = ((size_t)((n * 30 + h + ky) * 30) + (w + kx)) * 256 + k0;
      }
      const int4* g = (const int4*)(Bact + boff + part * 32);
      int4 d0 = g[0];
      int4 d1 = g[1];
      *(int4*)&Bt[i * LDS_STRIDE + part * 32 + 0]  = d0;
      *(int4*)&Bt[i * LDS_STRIDE + part * 32 + 16] = d1;
    }
    __syncthreads();

    v8i afrag[2];
#pragma unroll
    for (int mi = 0; mi < 2; ++mi) {
      const signed char* base = &At[(wm * 32 + mi * 16 + l15) * LDS_STRIDE + hi8];
      ((int2*)&afrag[mi])[0] = *(const int2*)(base + 0);
      ((int2*)&afrag[mi])[1] = *(const int2*)(base + 16);
      ((int2*)&afrag[mi])[2] = *(const int2*)(base + 32);
      ((int2*)&afrag[mi])[3] = *(const int2*)(base + 48);
    }
#pragma unroll
    for (int ni = 0; ni < 4; ++ni) {
      const signed char* base = &Bt[(wn * 64 + ni * 16 + l15) * LDS_STRIDE + hi8];
      v8i bfrag;
      ((int2*)&bfrag)[0] = *(const int2*)(base + 0);
      ((int2*)&bfrag)[1] = *(const int2*)(base + 16);
      ((int2*)&bfrag)[2] = *(const int2*)(base + 32);
      ((int2*)&bfrag)[3] = *(const int2*)(base + 48);
#pragma unroll
      for (int mi = 0; mi < 2; ++mi) {
        acc[mi][ni] = __builtin_amdgcn_wmma_i32_16x16x64_iu8(
            true, afrag[mi], true, bfrag, acc[mi][ni], false, false);
      }
    }
  }

#pragma unroll
  for (int mi = 0; mi < 2; ++mi) {
    const int mrow0 = mtile + wm * 32 + mi * 16 + hi8;
    float a0[8], b0[8];
#pragma unroll
    for (int r = 0; r < 8; ++r) { a0[r] = bnA[mrow0 + r]; b0[r] = bnB[mrow0 + r]; }

#pragma unroll
    for (int ni = 0; ni < 4; ++ni) {
      const int p = ntile + wn * 64 + ni * 16 + l15;
      if (MODE == 3) {
        const int n = p / 784, rem = p % 784;
#pragma unroll
        for (int r = 0; r < 8; ++r) {
          float v = rintf(a0[r] * (float)acc[mi][ni][r] + b0[r]);
          v = fminf(127.0f, fmaxf(-128.0f, v));
          const size_t oidx = ((size_t)n * 1024 + (mrow0 + r)) * 784 + rem;
          v += (float)xid[oidx];
          v = fmaxf(0.0f, fminf(127.0f, v));
          outf[oidx] = v;
        }
      } else {
        unsigned int lo = 0u, hiw = 0u;
#pragma unroll
        for (int r = 0; r < 8; ++r) {
          float v = rintf(a0[r] * (float)acc[mi][ni][r] + b0[r]);
          v = fmaxf(0.0f, fminf(127.0f, v));
          const unsigned int qb = ((unsigned int)(int)v) & 0xffu;
          if (r < 4) lo |= qb << (r * 8); else hiw |= qb << ((r - 4) * 8);
        }
        size_t off;
        if (MODE == 1) {
          const int n = p / 784, rem = p % 784, h = rem / 28, w = rem % 28;
          off = ((size_t)((n * 30 + h + 1) * 30) + (w + 1)) * 256 + mrow0;
        } else {
          off = (size_t)p * 256 + mrow0;
        }
        int2 st; st.x = (int)lo; st.y = (int)hiw;
        *(int2*)(out8 + off) = st;
      }
    }
  }
}

extern "C" void kernel_launch(void* const* d_in, const int* in_sizes, int n_in,
                              void* d_out, int out_size, void* d_ws, size_t ws_size,
                              hipStream_t stream) {
  const int*   x      = (const int*)d_in[0];
  const float* w2_1   = (const float*)d_in[1];
  const int*   s1     = (const int*)d_in[2];
  const float* w2_2   = (const float*)d_in[3];
  const int*   s2     = (const int*)d_in[4];
  const float* w2_3   = (const float*)d_in[5];
  const int*   s3     = (const int*)d_in[6];
  const float* alpha1 = (const float*)d_in[7];
  const float* beta1  = (const float*)d_in[8];
  const int*   q1     = (const int*)d_in[9];
  const float* alpha2 = (const float*)d_in[10];
  const float* beta2  = (const float*)d_in[11];
  const int*   q2     = (const int*)d_in[12];
  const float* alpha3 = (const float*)d_in[13];
  const float* beta3  = (const float*)d_in[14];
  const int*   q3     = (const int*)d_in[15];
  float* out = (float*)d_out;

  char* ws = (char*)d_ws;
  size_t off = 0;
  auto carve = [&](size_t bytes) { char* p = ws + off; off += (bytes + 255) & ~(size_t)255; return p; };
  signed char* x8    = (signed char*)carve(16u * 28 * 28 * 1024);
  signed char* w1    = (signed char*)carve(256u * 1024);
  signed char* wt    = (signed char*)carve(9u * 256 * 256);
  signed char* w3    = (signed char*)carve(1024u * 256);
  signed char* y1pad = (signed char*)carve(16u * 30 * 30 * 256);
  signed char* y2    = (signed char*)carve(12544u * 256);
  float* bnA1 = (float*)carve(256 * 4);
  float* bnB1 = (float*)carve(256 * 4);
  float* bnA2 = (float*)carve(256 * 4);
  float* bnB2 = (float*)carve(256 * 4);
  float* bnA3 = (float*)carve(1024 * 4);
  float* bnB3 = (float*)carve(1024 * 4);

  pack_x_kernel<<<50176, 256, 0, stream>>>(x, x8);
  pack_w11_kernel<<<1024, 256, 0, stream>>>(w2_1, s1, w1);
  pack_w33_kernel<<<2304, 256, 0, stream>>>(w2_2, s2, wt);
  pack_w11_kernel<<<1024, 256, 0, stream>>>(w2_3, s3, w3);
  prep_bn_kernel<<<1, 256, 0, stream>>>(alpha1, beta1, q1, bnA1, bnB1, 256);
  prep_bn_kernel<<<1, 256, 0, stream>>>(alpha2, beta2, q2, bnA2, bnB2, 256);
  prep_bn_kernel<<<4, 256, 0, stream>>>(alpha3, beta3, q3, bnA3, bnB3, 1024);
  zero16_kernel<<<900, 256, 0, stream>>>((int4*)y1pad);

  gemm_bn_kernel<1><<<dim3(98, 2), 256, 0, stream>>>(w1, x8, y1pad, nullptr, nullptr, bnA1, bnB1);
  gemm_bn_kernel<2><<<dim3(98, 2), 256, 0, stream>>>(wt, y1pad, y2, nullptr, nullptr, bnA2, bnB2);
  gemm_bn_kernel<3><<<dim3(98, 8), 256, 0, stream>>>(w3, y2, nullptr, out, x, bnA3, bnB3);
}
